// GroupedQueryAttention_19739669692842
// MI455X (gfx1250) — matrix operand order repaired and checked by reading, not compiled, not run on hardware
//
#include <hip/hip_runtime.h>
#include <hip/hip_bf16.h>

// ---------------------------------------------------------------------------
// GQA forward on gfx1250: bf16 WMMA (v_wmma_f32_16x16x32_bf16) everywhere.
// ---------------------------------------------------------------------------

typedef __attribute__((ext_vector_type(16))) __bf16 v16bf;
typedef __attribute__((ext_vector_type(8)))  float  v8f;

union Frag {
    v16bf v;
    uint4 q[2];
    unsigned short us[16];
};

__device__ __forceinline__ unsigned short f2bf(float f) {
    union { float f; unsigned int u; } x; x.f = f;
    unsigned int r = x.u + 0x7FFFu + ((x.u >> 16) & 1u);   // round-to-nearest-even
    return (unsigned short)(r >> 16);
}
__device__ __forceinline__ float bf2f(unsigned short h) {
    union { unsigned int u; float f; } x; x.u = ((unsigned int)h) << 16;
    return x.f;
}

__device__ __forceinline__ v8f wmma_bf16(v16bf a, v16bf b, v8f c) {
    return __builtin_amdgcn_wmma_f32_16x16x32_bf16(false, a, false, b,
                                                   (short)0, c, false, false);
}

__device__ __forceinline__ float half16_max(float v) {
#pragma unroll
    for (int m = 8; m >= 1; m >>= 1) v = fmaxf(v, __shfl_xor(v, m, 32));
    return v;
}

__device__ __forceinline__ void wait_lds() {
#if defined(__has_builtin) && __has_builtin(__builtin_amdgcn_s_wait_dscnt)
    __builtin_amdgcn_s_wait_dscnt(0);
#else
    asm volatile("s_wait_dscnt 0x0" ::: "memory");
#endif
}

// ---------------------------------------------------------------------------
// fp32 -> bf16 cast, 8 elements/thread (2x float4 in, 1x uint4 out)
// ---------------------------------------------------------------------------
__global__ void cast_f32_to_bf16_v8(const float4* __restrict__ in,
                                    uint4* __restrict__ out, long long n8) {
    long long i = (long long)blockIdx.x * blockDim.x + threadIdx.x;
    long long stride = (long long)gridDim.x * blockDim.x;
    for (; i < n8; i += stride) {
        float4 a = in[2 * i];
        float4 b = in[2 * i + 1];
        uint4 o;
        o.x = (unsigned)f2bf(a.x) | ((unsigned)f2bf(a.y) << 16);
        o.y = (unsigned)f2bf(a.z) | ((unsigned)f2bf(a.w) << 16);
        o.z = (unsigned)f2bf(b.x) | ((unsigned)f2bf(b.y) << 16);
        o.w = (unsigned)f2bf(b.z) | ((unsigned)f2bf(b.w) << 16);
        out[i] = o;
    }
}

// ---------------------------------------------------------------------------
// GEMM: Out[M,N] = A[M,K] * W[N,K]^T   (contraction over K, both row-major)
// mode 0: store fp32 row-major M x N
// mode 1: store bf16 head-major: ((b*nH+h)*Tdim+t)*64+d
// mode 2: store bf16 d-major (transposed): ((b*nH+h)*64+d)*Tdim+t
// Block: 256 thr = 8 waves; wave tile 32x32; block tile 64(M) x 128(N).
// ---------------------------------------------------------------------------
__global__ __launch_bounds__(256)
void gemm_bf16_wmma(const unsigned short* __restrict__ A,
                    const unsigned short* __restrict__ W,
                    void* __restrict__ Out,
                    int M, int N, int K, int mode, int Tdim, int nH) {
    const int lane   = threadIdx.x & 31;
    const int waveId = threadIdx.x >> 5;
    const int lane16 = lane & 15;
    const int kh     = lane >> 4;                 // which 16-lane half
    const int mBase  = blockIdx.y * 64  + (waveId >> 2) * 32;
    const int nBase  = blockIdx.x * 128 + (waveId & 3) * 32;

    v8f acc[2][2] = {};

    const unsigned short* arow0 = A + (size_t)(mBase      + lane16) * K;
    const unsigned short* arow1 = A + (size_t)(mBase + 16 + lane16) * K;
    const unsigned short* wrow0 = W + (size_t)(nBase      + lane16) * K;
    const unsigned short* wrow1 = W + (size_t)(nBase + 16 + lane16) * K;

    for (int k0 = 0; k0 < K; k0 += 32) {
        Frag a0, a1, b0, b1;
        // A frag, this code's own k order (B matches): lane half kh -> K = 16*kh + i, contiguous 16
        a0.q[0] = *(const uint4*)(arow0 + k0 + 16 * kh);
        a0.q[1] = *(const uint4*)(arow0 + k0 + 16 * kh + 8);
        a1.q[0] = *(const uint4*)(arow1 + k0 + 16 * kh);
        a1.q[1] = *(const uint4*)(arow1 + k0 + 16 * kh + 8);
        // B frag: lane = column; same k order as A: K = 16*kh + i, contiguous 16
        b0.q[0] = *(const uint4*)(wrow0 + k0 + 16 * kh);
        b0.q[1] = *(const uint4*)(wrow0 + k0 + 16 * kh + 8);
        b1.q[0] = *(const uint4*)(wrow1 + k0 + 16 * kh);
        b1.q[1] = *(const uint4*)(wrow1 + k0 + 16 * kh + 8);

        acc[0][0] = wmma_bf16(a0.v, b0.v, acc[0][0]);
        acc[0][1] = wmma_bf16(a0.v, b1.v, acc[0][1]);
        acc[1][0] = wmma_bf16(a1.v, b0.v, acc[1][0]);
        acc[1][1] = wmma_bf16(a1.v, b1.v, acc[1][1]);
    }

    // C/D layout: VGPR r, half kh -> row = r + 8*kh; col = lane16
#pragma unroll
    for (int mi = 0; mi < 2; ++mi)
#pragma unroll
        for (int ni = 0; ni < 2; ++ni)
#pragma unroll
            for (int r = 0; r < 8; ++r) {
                int row = mBase + mi * 16 + r + 8 * kh;
                int col = nBase + ni * 16 + lane16;
                float val = acc[mi][ni][r];
                if (mode == 0) {
                    ((float*)Out)[(size_t)row * N + col] = val;
                } else {
                    int b = row / Tdim, t = row % Tdim;
                    int h = col >> 6,  d = col & 63;
                    size_t idx = (mode == 1)
                        ? ((((size_t)(b * nH + h)) * Tdim + t) << 6) + d
                        : ((((size_t)(b * nH + h)) << 6) + d) * Tdim + t;
                    ((unsigned short*)Out)[idx] = f2bf(val);
                }
            }
}

// ---------------------------------------------------------------------------
// In-place RoPE on head-major (B, nH, T, 64) bf16; pairs (2*d2, 2*d2+1)
// Each thread processes one bf16 pair as a single u32.
// ---------------------------------------------------------------------------
__global__ void rope_inplace(unsigned short* __restrict__ t,
                             const float* __restrict__ cosb,
                             const float* __restrict__ sinb,
                             int T, long long total) {
    long long i = (long long)blockIdx.x * blockDim.x + threadIdx.x;
    long long stride = (long long)gridDim.x * blockDim.x;
    for (; i < total; i += stride) {
        int d2 = (int)(i & 31);
        long long rest = i >> 5;
        int tt = (int)(rest % T);
        long long bh = rest / T;
        size_t base = ((size_t)bh * T + tt) * 64 + 2 * d2;
        float c = cosb[tt * 32 + d2];
        float s = sinb[tt * 32 + d2];
        unsigned int pr = *(const unsigned int*)(t + base);
        float f0 = bf2f((unsigned short)(pr & 0xFFFFu));
        float f1 = bf2f((unsigned short)(pr >> 16));
        unsigned int o = (unsigned)f2bf(f0 * c - f1 * s)
                       | ((unsigned)f2bf(f0 * s + f1 * c) << 16);
        *(unsigned int*)(t + base) = o;
    }
}

// ---------------------------------------------------------------------------
// Flash attention: one wave per 32-row q tile; K-step 32; GQA 4:1.
// Q:  (B,32,T,64) bf16 head-major (roped)
// K:  (B,8,T,64)  bf16 head-major (roped)
// Vt: (B,8,64,T)  bf16 d-major (transposed) -> B-frags are contiguous loads
// Y:  (B*T, 2048) bf16
// Softmax is done base-2 (v_exp_f32 is exp2); scale 0.125*log2(e) is folded
// into the Q fragments once. Row sums come from a P x ones WMMA (C layout
// lands each row's sum in the lanes owning that row's running stats).
// ---------------------------------------------------------------------------
__global__ __launch_bounds__(128)
void attn_fwd(const unsigned short* __restrict__ Q,
              const unsigned short* __restrict__ Kc,
              const unsigned short* __restrict__ Vt,
              unsigned short* __restrict__ Y, int T) {
    const int lane   = threadIdx.x & 31;
    const int w      = threadIdx.x >> 5;
    const int lane16 = lane & 15;
    const int kh     = lane >> 4;
    const int qt     = blockIdx.x * 4 + w;        // q tile of 32 rows
    const int qbase  = qt * 32;
    const int bh     = blockIdx.y;                // b*32 + h
    const int b      = bh >> 5;
    const int h      = bh & 31;
    const int kvh    = h >> 2;                    // 4:1 GQA

    const unsigned short* Qh = Q  + ((size_t)bh * T) * 64;
    const unsigned short* Kh = Kc + ((size_t)(b * 8 + kvh) * T) * 64;
    const unsigned short* Vh = Vt + ((size_t)(b * 8 + kvh) * 64) * T;

    __shared__ unsigned short P[4][32][32];       // per-wave P transpose buffer

    const float SCALE_LOG2 = 0.125f * 1.44269504088896340736f; // 1/sqrt(64)*log2(e)

    // Q fragments (A layout): 2 row-subtiles x 2 d-halves; scale folded in
    Frag aq[2][2];
#pragma unroll
    for (int qs = 0; qs < 2; ++qs) {
        const unsigned short* qr = Qh + (size_t)(qbase + qs * 16 + lane16) * 64;
        aq[qs][0].q[0] = *(const uint4*)(qr + 16 * kh);
        aq[qs][0].q[1] = *(const uint4*)(qr + 16 * kh + 8);
        aq[qs][1].q[0] = *(const uint4*)(qr + 32 + 16 * kh);
        aq[qs][1].q[1] = *(const uint4*)(qr + 32 + 16 * kh + 8);
#pragma unroll
        for (int dh = 0; dh < 2; ++dh)
#pragma unroll
            for (int i = 0; i < 16; ++i)
                aq[qs][dh].us[i] = f2bf(bf2f(aq[qs][dh].us[i]) * SCALE_LOG2);
    }

    Frag ones;                                    // B-frag of 1.0 bf16
#pragma unroll
    for (int i = 0; i < 16; ++i) ones.us[i] = 0x3F80;

    float mrun[2][8], lrun[2][8];
#pragma unroll
    for (int qs = 0; qs < 2; ++qs)
#pragma unroll
        for (int r = 0; r < 8; ++r) { mrun[qs][r] = -1e30f; lrun[qs][r] = 0.f; }
    v8f o[2][4] = {};

    const int qmaxg = qbase + 31;
    for (int k32 = 0; k32 <= qmaxg; k32 += 32) {
        // --- K fragments (B layout): lane = score column (= key row) ---
        const unsigned short* kr0 = Kh + (size_t)(k32 + lane16) * 64;
        const unsigned short* kr1 = Kh + (size_t)(k32 + 16 + lane16) * 64;
        Frag bk[2][2];
        bk[0][0].q[0] = *(const uint4*)(kr0 + 16 * kh);
        bk[0][0].q[1] = *(const uint4*)(kr0 + 16 * kh + 8);
        bk[0][1].q[0] = *(const uint4*)(kr0 + 32 + 16 * kh);
        bk[0][1].q[1] = *(const uint4*)(kr0 + 32 + 16 * kh + 8);
        bk[1][0].q[0] = *(const uint4*)(kr1 + 16 * kh);
        bk[1][0].q[1] = *(const uint4*)(kr1 + 16 * kh + 8);
        bk[1][1].q[0] = *(const uint4*)(kr1 + 32 + 16 * kh);
        bk[1][1].q[1] = *(const uint4*)(kr1 + 32 + 16 * kh + 8);

        // --- scores (log2 domain): 2 q-subtiles x 2 n-subtiles -> 8 WMMAs ---
        v8f s[2][2] = {};
#pragma unroll
        for (int qs = 0; qs < 2; ++qs)
#pragma unroll
            for (int nt = 0; nt < 2; ++nt) {
                s[qs][nt] = wmma_bf16(aq[qs][0].v, bk[nt][0].v, s[qs][nt]);
                s[qs][nt] = wmma_bf16(aq[qs][1].v, bk[nt][1].v, s[qs][nt]);
            }

        // --- online softmax (base 2) per q-subtile ---
        float facs[2][8];
#pragma unroll
        for (int qs = 0; qs < 2; ++qs) {
            const bool needMask = (k32 + 31) > (qbase + qs * 16);
#pragma unroll
            for (int r = 0; r < 8; ++r) {
                int mg = qbase + qs * 16 + r + 8 * kh;
                float v0 = s[qs][0][r];
                float v1 = s[qs][1][r];
                if (needMask) {
                    int n0 = k32 + lane16;
                    if (n0 > mg)      v0 = -1e30f;     // causal mask
                    if (n0 + 16 > mg) v1 = -1e30f;
                }
                float tmax = half16_max(fmaxf(v0, v1));
                float nm   = fmaxf(mrun[qs][r], tmax);
                float fac  = exp2f(mrun[qs][r] - nm);
                float p0   = exp2f(v0 - nm);
                float p1   = exp2f(v1 - nm);
                mrun[qs][r] = nm;
                facs[qs][r] = fac;
                o[qs][0][r] *= fac; o[qs][1][r] *= fac;
                o[qs][2][r] *= fac; o[qs][3][r] *= fac;
                P[w][qs * 16 + r + 8 * kh][lane16]      = f2bf(p0);
                P[w][qs * 16 + r + 8 * kh][lane16 + 16] = f2bf(p1);
            }
        }
        wait_lds();                                // per-wave LDS transpose fence

        Frag pa[2];                                // P as A fragments (16x32 each)
#pragma unroll
        for (int qs = 0; qs < 2; ++qs) {
            pa[qs].q[0] = *(const uint4*)&P[w][qs * 16 + lane16][16 * kh];
            pa[qs].q[1] = *(const uint4*)&P[w][qs * 16 + lane16][16 * kh + 8];
        }

        // --- row sums via P x ones (replaces 64 shuffle ops per step) ---
        v8f rs[2] = {};
        rs[0] = wmma_bf16(pa[0].v, ones.v, rs[0]);
        rs[1] = wmma_bf16(pa[1].v, ones.v, rs[1]);
#pragma unroll
        for (int qs = 0; qs < 2; ++qs)
#pragma unroll
            for (int r = 0; r < 8; ++r)
                lrun[qs][r] = lrun[qs][r] * facs[qs][r] + rs[qs][r];

        // --- V^T fragments (B layout): row d = ni*16+lane16, contiguous over t ---
#pragma unroll
        for (int ni = 0; ni < 4; ++ni) {
            const unsigned short* vr = Vh + (size_t)(ni * 16 + lane16) * T + k32 + 16 * kh;
            Frag bv;
            bv.q[0] = *(const uint4*)(vr);
            bv.q[1] = *(const uint4*)(vr + 8);
            o[0][ni] = wmma_bf16(pa[0].v, bv.v, o[0][ni]);
            o[1][ni] = wmma_bf16(pa[1].v, bv.v, o[1][ni]);
        }
    }

    // normalize + store to (B*T, 2048) bf16 for the wo GEMM
#pragma unroll
    for (int qs = 0; qs < 2; ++qs)
#pragma unroll
        for (int r = 0; r < 8; ++r) {
            float inv = 1.0f / lrun[qs][r];
            int tg = qbase + qs * 16 + r + 8 * kh;
#pragma unroll
            for (int ni = 0; ni < 4; ++ni) {
                float val = o[qs][ni][r] * inv;
                Y[((size_t)b * T + tg) * 2048 + h * 64 + ni * 16 + lane16] = f2bf(val);
            }
        }
}

// ---------------------------------------------------------------------------
extern "C" void kernel_launch(void* const* d_in, const int* in_sizes, int n_in,
                              void* d_out, int out_size, void* d_ws, size_t ws_size,
                              hipStream_t stream) {
    const float* x    = (const float*)d_in[0];
    const float* cosb = (const float*)d_in[1];
    const float* sinb = (const float*)d_in[2];
    const float* wq   = (const float*)d_in[3];
    const float* wk   = (const float*)d_in[4];
    const float* wv   = (const float*)d_in[5];
    const float* wo   = (const float*)d_in[6];

    constexpr int Bc = 2, Tc = 2048, DIM = 2048, H = 32, KVH = 8, D = 64;
    constexpr long long MT = (long long)Bc * Tc;          // 4096 rows
    constexpr int NKV = KVH * D;                          // 512

    char* ws = (char*)d_ws;
    auto alloc = [&](size_t bytes) {
        char* p = ws;
        ws += (bytes + 255) & ~(size_t)255;
        return p;
    };
    unsigned short* x_bf  = (unsigned short*)alloc((size_t)MT * DIM * 2);
    unsigned short* wq_bf = (unsigned short*)alloc((size_t)DIM * DIM * 2);
    unsigned short* wk_bf = (unsigned short*)alloc((size_t)NKV * DIM * 2);
    unsigned short* wv_bf = (unsigned short*)alloc((size_t)NKV * DIM * 2);
    unsigned short* wo_bf = (unsigned short*)alloc((size_t)DIM * DIM * 2);
    unsigned short* q_t   = (unsigned short*)alloc((size_t)Bc * H   * Tc * D * 2);
    unsigned short* k_t   = (unsigned short*)alloc((size_t)Bc * KVH * Tc * D * 2);
    unsigned short* v_dt  = (unsigned short*)alloc((size_t)Bc * KVH * D * Tc * 2);
    unsigned short* y_bf  = (unsigned short*)alloc((size_t)MT * DIM * 2);
    (void)ws_size; (void)in_sizes; (void)n_in; (void)out_size;

    auto cast = [&](const float* src, unsigned short* dst, long long n) {
        long long n8 = n / 8;
        long long blocks = (n8 + 255) / 256;
        if (blocks > 2048) blocks = 2048;
        cast_f32_to_bf16_v8<<<dim3((unsigned)blocks), dim3(256), 0, stream>>>(
            (const float4*)src, (uint4*)dst, n8);
    };
    cast(x,  x_bf,  MT * DIM);
    cast(wq, wq_bf, (long long)DIM * DIM);
    cast(wk, wk_bf, (long long)NKV * DIM);
    cast(wv, wv_bf, (long long)NKV * DIM);
    cast(wo, wo_bf, (long long)DIM * DIM);

    // q/k projections -> head-major bf16; v projection -> d-major (transposed)
    gemm_bf16_wmma<<<dim3(DIM / 128, MT / 64), 256, 0, stream>>>(
        x_bf, wq_bf, q_t, (int)MT, DIM, DIM, 1, Tc, H);
    gemm_bf16_wmma<<<dim3(NKV / 128, MT / 64), 256, 0, stream>>>(
        x_bf, wk_bf, k_t, (int)MT, NKV, DIM, 1, Tc, KVH);
    gemm_bf16_wmma<<<dim3(NKV / 128, MT / 64), 256, 0, stream>>>(
        x_bf, wv_bf, v_dt, (int)MT, NKV, DIM, 2, Tc, KVH);

    // RoPE in place on q and k
    {
        long long tq = (long long)Bc * H * Tc * 32;
        rope_inplace<<<dim3(4096), dim3(256), 0, stream>>>(q_t, cosb, sinb, Tc, tq);
        long long tk = (long long)Bc * KVH * Tc * 32;
        rope_inplace<<<dim3(2048), dim3(256), 0, stream>>>(k_t, cosb, sinb, Tc, tk);
    }

    // flash attention -> y_bf (B*T, 2048); 32 q rows per wave, 4 waves/block
    attn_fwd<<<dim3(Tc / 128, Bc * H), 128, 0, stream>>>(q_t, k_t, v_dt, y_bf, Tc);

    // output projection -> fp32 d_out
    gemm_bf16_wmma<<<dim3(DIM / 128, MT / 64), 256, 0, stream>>>(
        y_bf, wo_bf, d_out, (int)MT, DIM, DIM, 0, 0, 0);
}
